// BsPINN_45938970198305
// MI455X (gfx1250) — matrix operand order repaired and checked by reading, not compiled, not run on hardware
//
#include <hip/hip_runtime.h>
#include <hip/hip_bf16.h>

typedef __attribute__((ext_vector_type(16))) _Float16 v16h;
typedef __attribute__((ext_vector_type(8)))  float    v8f;

#define PITCH   520            // halves per LDS row: 1040B, 16B-aligned, bank-staggered
#define NT_DOM  4096           // 65536 / 16 domain tiles
#define NT_BND  1024           // 16384 / 16 boundary tiles
#define WH_BYTES (4*512*512*2) // 4 hidden-layer f16 weight mats

union FragU { v16h h; uint4 u[2]; };

// ---------------------------------------------------------------------------
// Kernel 1: convert W1..W4 to f16 with block-diagonal masks baked in (N-major,
// matching the WMMA B-fragment layout: lane<->N column, contiguous K halves),
// plus f16 copy of W5.
// ---------------------------------------------------------------------------
__global__ __launch_bounds__(256) void bspinn_prep(
    const float* __restrict__ W1, const float* __restrict__ W2,
    const float* __restrict__ W3, const float* __restrict__ W4,
    const float* __restrict__ W5,
    _Float16* __restrict__ Wh, _Float16* __restrict__ W5h)
{
  int idx = blockIdx.x * 256 + threadIdx.x;
  if (idx < 4 * 512 * 512) {
    int L = idx >> 18;          // 0..3  <-> python weight index l = L+1
    int r = (idx >> 9) & 511;   // K (input) index
    int c = idx & 511;          // N (output) index
    const float* W = (L == 0) ? W1 : (L == 1) ? W2 : (L == 2) ? W3 : W4;
    float w = W[(r << 9) + c];
    if (L >= 1) {               // masks on l=2,3,4: square diag blocks of 256/128/64
      int bs = 512 >> L;
      if ((r / bs) != (c / bs)) w = 0.0f;
    }
    Wh[(L << 18) + (c << 9) + r] = (_Float16)w;
  } else if (idx < 4 * 512 * 512 + 512) {
    int k = idx - 4 * 512 * 512;
    W5h[k] = (_Float16)W5[k];
  }
}

__global__ void bspinn_init(float* sums) {
  if (threadIdx.x < 2) sums[threadIdx.x] = 0.0f;
}

// ---------------------------------------------------------------------------
// Kernel 2: fused PINN forward + Taylor-mode 2nd-derivative propagation.
// One workgroup (8 wave32) per 16-point tile. NCH is a compile-time channel
// count (5 = domain: z, zx, zy, zxx, zyy; 1 = boundary: z only) so the
// accumulator array is fully register-resident (no runtime indexing).
// ---------------------------------------------------------------------------
template <int NCH>
__global__ __launch_bounds__(256) void bspinn_main(
    const float* __restrict__ X,
    const float* __restrict__ W0, const float* __restrict__ b0,
    const float* __restrict__ b1, const float* __restrict__ b2,
    const float* __restrict__ b3, const float* __restrict__ b4,
    const float* __restrict__ b5,
    const _Float16* __restrict__ Wh, const _Float16* __restrict__ W5h,
    float* __restrict__ sums, int pointBase)
{
  extern __shared__ char smem_raw[];
  _Float16* act  = (_Float16*)smem_raw;            // [NCH][16][PITCH] halves
  float*    pts  = (float*)(smem_raw + NCH * 16 * PITCH * 2);  // [16][2]
  float*    dots = pts + 32;                                   // [NCH][16]
  float*    red  = dots + NCH * 16;                            // [16]

  const int tid  = threadIdx.x;
  const int lane = tid & 31;
  const int wave = tid >> 5;
  const int pBase = pointBase + blockIdx.x * 16;
  constexpr bool domain = (NCH == 5);

  // stage the 16 points (x,y)
  if (tid < 32) pts[tid] = X[pBase * 2 + tid];
  __syncthreads();

  // ---- layer 0: 2 -> 512 (tiny, scalar) with input normalization 2x-1 ----
  for (int e = tid; e < 16 * 512; e += 256) {
    int m = e >> 9, j = e & 511;
    float xh = 2.0f * pts[2 * m]     - 1.0f;
    float yh = 2.0f * pts[2 * m + 1] - 1.0f;
    float w0 = W0[j], w1 = W0[512 + j];
    float s  = xh * w0 + yh * w1 + b0[j];
    float sn = __sinf(s), cs = __cosf(s);
    act[(0 * 16 + m) * PITCH + j] = (_Float16)sn;
    if (domain) {
      float sx = 2.0f * w0, sy = 2.0f * w1;        // d(pre)/dx, /dy; 2nd deriv = 0
      act[(1 * 16 + m) * PITCH + j] = (_Float16)(cs * sx);
      act[(2 * 16 + m) * PITCH + j] = (_Float16)(cs * sy);
      act[(3 * 16 + m) * PITCH + j] = (_Float16)(-sn * sx * sx);
      act[(4 * 16 + m) * PITCH + j] = (_Float16)(-sn * sy * sy);
    }
  }
  __syncthreads();

  const int colBase = wave * 64;          // this wave owns 4 N-tiles of 16
  const int mRow    = (lane >> 4) * 8;    // C/D layout: M = vgpr + 8*hi
  const int nCol    = lane & 15;          // C/D layout: N = lane&15
  const int aRow    = lane & 15;          // A layout: row = lane&15
  const int aKoff   = (lane >> 4) * 8;    // A layout: hi half selects K subset
  const int bK      = (lane & 15) + ((lane >> 4) << 4);  // unused; B loads map lane <-> N (nCol)

  // ---- 4 hidden layers: [NCH*16 x 512] @ [512 x 512] via WMMA ----
  for (int L = 0; L < 4; ++L) {
    const _Float16* W = Wh + (size_t)L * 512 * 512;
    v8f acc[NCH][4];
    const v8f vzero = { 0.f, 0.f, 0.f, 0.f, 0.f, 0.f, 0.f, 0.f };
#pragma unroll
    for (int c = 0; c < NCH; ++c)
#pragma unroll
      for (int t = 0; t < 4; ++t) acc[c][t] = vzero;

#pragma unroll 4
    for (int kt = 0; kt < 16; ++kt) {
      const int k0 = kt * 32;
      // B fragments: lane holds column n = lane&15 (N-major Wh), K chunks at k0+hi*8 and k0+16+hi*8
      FragU Bf[4];
#pragma unroll
      for (int t = 0; t < 4; ++t) {
        const uint4* p = (const uint4*)(W + (size_t)(colBase + t * 16 + nCol) * 512 + k0 + aKoff);
        Bf[t].u[0] = p[0];
        Bf[t].u[1] = p[2];
      }
      // A fragments per channel: two 16B chunks at k0+hi*8 and k0+16+hi*8
#pragma unroll
      for (int c = 0; c < NCH; ++c) {
        const _Float16* ar = act + (c * 16 + aRow) * PITCH;
        FragU Af;
        Af.u[0] = *(const uint4*)(ar + k0 + aKoff);
        Af.u[1] = *(const uint4*)(ar + k0 + 16 + aKoff);
#pragma unroll
        for (int t = 0; t < 4; ++t) {
          acc[c][t] = __builtin_amdgcn_wmma_f32_16x16x32_f16(
              false, Af.h, false, Bf[t].h, (short)0, acc[c][t], false, false);
        }
      }
    }
    __syncthreads();   // all waves finished reading act for this layer

    // epilogue: sine chain rule in f32, write next activations (f16) to LDS
    const float* bl = (L == 0) ? b1 : (L == 1) ? b2 : (L == 2) ? b3 : b4;
#pragma unroll
    for (int t = 0; t < 4; ++t) {
      const int n  = colBase + t * 16 + nCol;
      const float bn = bl[n];
#pragma unroll
      for (int v = 0; v < 8; ++v) {
        const int m = mRow + v;
        float z  = acc[0][t][v] + bn;      // bias only on value channel
        float sn = __sinf(z), cs = __cosf(z);
        act[(0 * 16 + m) * PITCH + n] = (_Float16)sn;
        if (domain) {
          float zx  = acc[1 % NCH][t][v], zy  = acc[2 % NCH][t][v];
          float zxx = acc[3 % NCH][t][v], zyy = acc[4 % NCH][t][v];
          act[(1 * 16 + m) * PITCH + n] = (_Float16)(cs * zx);
          act[(2 * 16 + m) * PITCH + n] = (_Float16)(cs * zy);
          act[(3 * 16 + m) * PITCH + n] = (_Float16)(cs * zxx - sn * zx * zx);
          act[(4 * 16 + m) * PITCH + n] = (_Float16)(cs * zyy - sn * zy * zy);
        }
      }
    }
    __syncthreads();
  }

  // ---- final layer: 512 -> 1 dot products per (channel, point) ----
  if (tid < NCH * 16) {
    int c = tid >> 4, m = tid & 15;
    const _Float16* ar = act + (c * 16 + m) * PITCH;
    float s = 0.0f;
    for (int k = 0; k < 512; ++k) s += (float)ar[k] * (float)W5h[k];
    dots[c * 16 + m] = s;
  }
  __syncthreads();

  // ---- residual / boundary term + block reduction ----
  if (tid < 16) {
    const int m = tid;
    float u = dots[m] + b5[0];
    float val;
    if (domain) {
      float uxx = dots[(3 % NCH) * 16 + m], uyy = dots[(4 % NCH) * 16 + m];
      float x = pts[2 * m], y = pts[2 * m + 1];
      float f  = 64.0f * __sinf(8.0f * x) * __sinf(8.0f * y);   // K0^2 sin(K0 x) sin(K0 y)
      float eq = -uxx - uyy - 64.0f * u - f;
      val = eq * eq;
    } else {
      val = u * u;
    }
    red[m] = val;
  }
  __syncthreads();
  if (tid == 0) {
    float s = 0.0f;
#pragma unroll
    for (int m = 0; m < 16; ++m) s += red[m];
    atomicAdd(&sums[domain ? 0 : 1], s);
  }
}

__global__ void bspinn_final(const float* __restrict__ sums, float* __restrict__ out) {
  if (threadIdx.x == 0)
    out[0] = sums[0] * (1.0f / 65536.0f) + 100.0f * sums[1] * (1.0f / 16384.0f);
}

// ---------------------------------------------------------------------------
extern "C" void kernel_launch(void* const* d_in, const int* in_sizes, int n_in,
                              void* d_out, int out_size, void* d_ws, size_t ws_size,
                              hipStream_t stream) {
  const float* X  = (const float*)d_in[0];
  const float* W0 = (const float*)d_in[1];
  const float* b0 = (const float*)d_in[2];
  const float* W1 = (const float*)d_in[3];
  const float* b1 = (const float*)d_in[4];
  const float* W2 = (const float*)d_in[5];
  const float* b2 = (const float*)d_in[6];
  const float* W3 = (const float*)d_in[7];
  const float* b3 = (const float*)d_in[8];
  const float* W4 = (const float*)d_in[9];
  const float* b4 = (const float*)d_in[10];
  const float* W5 = (const float*)d_in[11];
  const float* b5 = (const float*)d_in[12];

  char* ws = (char*)d_ws;
  _Float16* Wh   = (_Float16*)ws;                      // 2 MB: masked f16 W1..W4 (N-major)
  _Float16* W5h  = (_Float16*)(ws + WH_BYTES);         // 1 KB
  float*    sums = (float*)(ws + WH_BYTES + 1024);     // 2 floats

  bspinn_prep<<<4098, 256, 0, stream>>>(W1, W2, W3, W4, W5, Wh, W5h);
  bspinn_init<<<1, 32, 0, stream>>>(sums);

  // domain: 5 channels, LDS = 5*16*PITCH*2 + pts/dots/red
  const size_t lds_dom = (size_t)5 * 16 * PITCH * 2 + (32 + 5 * 16 + 16) * 4;
  bspinn_main<5><<<NT_DOM, 256, lds_dom, stream>>>(
      X, W0, b0, b1, b2, b3, b4, b5, Wh, W5h, sums, 0);

  // boundary: 1 channel
  const size_t lds_bnd = (size_t)1 * 16 * PITCH * 2 + (32 + 1 * 16 + 16) * 4;
  bspinn_main<1><<<NT_BND, 256, lds_bnd, stream>>>(
      X, W0, b0, b1, b2, b3, b4, b5, Wh, W5h, sums, 65536);

  bspinn_final<<<1, 32, 0, stream>>>(sums, (float*)d_out);
}
